// MambaFull_57801669869692
// MI455X (gfx1250) — matrix operand order repaired and checked by reading, not compiled, not run on hardware
//
#include <hip/hip_runtime.h>

#define DMm 512
#define DIi 1024
#define DST 16
#define DRr 32
#define NLl 4
#define BSZ 16
#define SEQL 200
#define CCc 200
#define NTOK (BSZ * SEQL) /* 3200 */

typedef __attribute__((ext_vector_type(16))) _Float16 v16h;
typedef __attribute__((ext_vector_type(8))) float v8f;

union V16H { v16h v; unsigned int u[8]; };
union V8F  { v8f  v; float f[8]; };

__device__ __forceinline__ unsigned short f2h(float x) {
  _Float16 h = (_Float16)x;
  return __builtin_bit_cast(unsigned short, h);
}
__device__ __forceinline__ float sigmoidf_(float x) { return 1.0f / (1.0f + __expf(-x)); }
__device__ __forceinline__ float siluf_(float x) { return x * sigmoidf_(x); }
__device__ __forceinline__ float softplusf_(float x) { return (x > 20.0f) ? x : log1pf(__expf(x)); }

// ---------------------------------------------------------------------------
// Generic fp32 -> fp16 conversion (weights converted once per launch)
// ---------------------------------------------------------------------------
__global__ __launch_bounds__(256) void cvt_f32_f16(const float* __restrict__ src,
                                                   unsigned short* __restrict__ dst, int count) {
  int i = blockIdx.x * 256 + threadIdx.x;
  if (i < count) dst[i] = f2h(src[i]);
}

// ---------------------------------------------------------------------------
// Embedding: h[n,d] = x[n,0]*W_emb[d,0] + x[n,1]*W_emb[d,1] + b_emb[d]
// ---------------------------------------------------------------------------
__global__ __launch_bounds__(256) void embed_kernel(const float* __restrict__ x,
                                                    const float* __restrict__ We,
                                                    const float* __restrict__ be,
                                                    float* __restrict__ h) {
  int idx = blockIdx.x * 256 + threadIdx.x;
  if (idx >= NTOK * DMm) return;
  int n = idx / DMm, d = idx % DMm;
  h[idx] = x[n * 2 + 0] * We[d * 2 + 0] + x[n * 2 + 1] * We[d * 2 + 1] + be[d];
}

// ---------------------------------------------------------------------------
// residual (+=h when add_h) then LayerNorm -> f16 activation for next GEMM
// ---------------------------------------------------------------------------
__global__ __launch_bounds__(256) void ln_kernel(float* __restrict__ res, const float* __restrict__ h,
                                                 int add_h, const float* __restrict__ w,
                                                 const float* __restrict__ b,
                                                 unsigned short* __restrict__ outh) {
  __shared__ float sred[256];
  const int n = blockIdx.x, tid = threadIdx.x;
  const size_t base = (size_t)n * DMm;
  float v0 = h[base + tid];
  float v1 = h[base + 256 + tid];
  if (add_h) { v0 += res[base + tid]; v1 += res[base + 256 + tid]; }
  res[base + tid] = v0;
  res[base + 256 + tid] = v1;
  sred[tid] = v0 + v1;
  __syncthreads();
  for (int s = 128; s > 0; s >>= 1) { if (tid < s) sred[tid] += sred[tid + s]; __syncthreads(); }
  float mean = sred[0] * (1.0f / DMm);
  __syncthreads();
  float d0 = v0 - mean, d1 = v1 - mean;
  sred[tid] = d0 * d0 + d1 * d1;
  __syncthreads();
  for (int s = 128; s > 0; s >>= 1) { if (tid < s) sred[tid] += sred[tid + s]; __syncthreads(); }
  float rstd = rsqrtf(sred[0] * (1.0f / DMm) + 1e-5f);
  outh[base + tid]       = f2h(d0 * rstd * w[tid] + b[tid]);
  outh[base + 256 + tid] = f2h(d1 * rstd * w[tid + 256] + b[tid + 256]);
}

// ---------------------------------------------------------------------------
// final residual add + RMSNorm -> f16 for head GEMM
// ---------------------------------------------------------------------------
__global__ __launch_bounds__(256) void rms_final_kernel(const float* __restrict__ res,
                                                        const float* __restrict__ h,
                                                        const float* __restrict__ w,
                                                        const float* __restrict__ b,
                                                        unsigned short* __restrict__ outh) {
  __shared__ float sred[256];
  const int n = blockIdx.x, tid = threadIdx.x;
  const size_t base = (size_t)n * DMm;
  float v0 = res[base + tid] + h[base + tid];
  float v1 = res[base + 256 + tid] + h[base + 256 + tid];
  sred[tid] = v0 * v0 + v1 * v1;
  __syncthreads();
  for (int s = 128; s > 0; s >>= 1) { if (tid < s) sred[tid] += sred[tid + s]; __syncthreads(); }
  float rr = rsqrtf(sred[0] * (1.0f / DMm) + 1e-5f);
  outh[base + tid]       = f2h(v0 * rr * w[tid] + b[tid]);
  outh[base + 256 + tid] = f2h(v1 * rr * w[tid + 256] + b[tid + 256]);
}

// ---------------------------------------------------------------------------
// WMMA GEMM: Out[N,M] = A[N,K] @ W[M,K]^T         (A, W f16 as packed u32)
// Block: 256 thr = 8 waves; block tile 64 rows x 64 cols; K-step 64.
// Wave w: row tile rw=w&3 (16 rows), col strip cw=w>>2 (32 cols = 2 WMMA tiles)
// -> 4 v_wmma per barrier interval per wave, A operand reused across 2 tiles.
// Software pipelined: next K-slab is loaded to registers (global_load_b128)
// during the current slab's WMMA work, committed to LDS between barriers.
// LDS stride 36 uints: 16B aligned for b128 stores AND conflict-free gathers.
// ---------------------------------------------------------------------------
__global__ __launch_bounds__(256) void gemm_f16_wmma(const unsigned int* __restrict__ A,
                                                     const unsigned int* __restrict__ W,
                                                     float* __restrict__ Out,
                                                     int N, int K, int M) {
  __shared__ __attribute__((aligned(16))) unsigned int As[64 * 36];
  __shared__ __attribute__((aligned(16))) unsigned int Ws[64 * 36];
  (void)N;
  const int tid = threadIdx.x;
  const int lane = tid & 31;
  const int wave = tid >> 5;
  const int rw = wave & 3;   // row tile 0..3
  const int cw = wave >> 2;  // col strip 0..1
  const int n0 = blockIdx.y * 64;
  const int m0 = blockIdx.x * 64;
  const int Kp = K >> 1;  // u32 per row

  // copy mapping: 512 uint4 quads per slab; thread t owns quads t and t+256
  const int cr0 = tid >> 3;            // rows 0..31
  const int cr1 = cr0 + 32;            // rows 32..63
  const int cj  = (tid & 7) * 4;       // uint offset within 32-uint slab row

  const uint4 zero4 = make_uint4(0u, 0u, 0u, 0u);
  auto ldA = [&](int k0, uint4& x0, uint4& x1) {
    const unsigned int* base0 = A + (size_t)(n0 + cr0) * Kp + (k0 >> 1) + cj;
    const unsigned int* base1 = A + (size_t)(n0 + cr1) * Kp + (k0 >> 1) + cj;
    x0 = *(const uint4*)base0;
    x1 = *(const uint4*)base1;
    __builtin_prefetch(base0 + 64, 0, 1);  // two slabs ahead -> global_prefetch
  };
  auto ldW = [&](int k0, uint4& x0, uint4& x1) {
    x0 = (m0 + cr0 < M) ? *(const uint4*)(W + (size_t)(m0 + cr0) * Kp + (k0 >> 1) + cj) : zero4;
    x1 = (m0 + cr1 < M) ? *(const uint4*)(W + (size_t)(m0 + cr1) * Kp + (k0 >> 1) + cj) : zero4;
  };
  auto commit = [&](const uint4& a0, const uint4& a1, const uint4& w0, const uint4& w1) {
    *(uint4*)&As[cr0 * 36 + cj] = a0;
    *(uint4*)&As[cr1 * 36 + cj] = a1;
    *(uint4*)&Ws[cr0 * 36 + cj] = w0;
    *(uint4*)&Ws[cr1 * 36 + cj] = w1;
  };

  V8F acc0, acc1;
#pragma unroll
  for (int r = 0; r < 8; ++r) { acc0.f[r] = 0.0f; acc1.f[r] = 0.0f; }

  const int arow  = rw * 16 + (lane & 15);
  const int ah    = (lane < 16) ? 0 : 8;  // +16 halves (K) for upper lane group
  const int brow0 = cw * 32 + (lane & 15);
  const int bh    = (lane < 16) ? 0 : 8;  // +16 halves (K) for upper lane group

  uint4 a0, a1, w0, w1;
  ldA(0, a0, a1);
  ldW(0, w0, w1);
  commit(a0, a1, w0, w1);
  __syncthreads();

  for (int k0 = 0;;) {
    const int kn = k0 + 64;
    const bool more = kn < K;
    uint4 na0, na1, nw0, nw1;
    if (more) { ldA(kn, na0, na1); ldW(kn, nw0, nw1); }

    // 2 K-substeps x 2 col tiles = 4 WMMA; A operand reused across col tiles
#pragma unroll
    for (int s = 0; s < 2; ++s) {
      V16H a, b0, b1;
      const unsigned int* ap  = &As[arow * 36 + s * 16 + ah];
      const unsigned int* bp0 = &Ws[brow0 * 36 + s * 16 + bh];
      const unsigned int* bp1 = bp0 + 16 * 36;
#pragma unroll
      for (int r = 0; r < 8; ++r) {
        a.u[r]  = ap[r];
        b0.u[r] = bp0[r];
        b1.u[r] = bp1[r];
      }
      acc0.v = __builtin_amdgcn_wmma_f32_16x16x32_f16(false, a.v, false, b0.v,
                                                      (short)0, acc0.v, false, false);
      acc1.v = __builtin_amdgcn_wmma_f32_16x16x32_f16(false, a.v, false, b1.v,
                                                      (short)0, acc1.v, false, false);
    }

    if (!more) break;
    __syncthreads();
    commit(na0, na1, nw0, nw1);
    __syncthreads();
    k0 = kn;
  }

  // D layout: vgpr r, lanes 0-15 -> M=r, lanes 16-31 -> M=r+8; N = lane&15
  const int ocol = m0 + cw * 32 + (lane & 15);
  const int rbase = n0 + rw * 16 + ((lane < 16) ? 0 : 8);
#pragma unroll
  for (int r = 0; r < 8; ++r) {
    if (ocol < M)      Out[(size_t)(rbase + r) * M + ocol]      = acc0.f[r];
    if (ocol + 16 < M) Out[(size_t)(rbase + r) * M + ocol + 16] = acc1.f[r];
  }
}

// ---------------------------------------------------------------------------
// Causal depthwise conv (k=4) + SiLU -> u (f32 for scan) and u_h (f16 for GEMM)
// ---------------------------------------------------------------------------
__global__ __launch_bounds__(256) void conv_silu_kernel(const float* __restrict__ xz,
                                                        const float* __restrict__ cw,
                                                        const float* __restrict__ cb,
                                                        float* __restrict__ u,
                                                        unsigned short* __restrict__ u_h) {
  int idx = blockIdx.x * 256 + threadIdx.x;
  if (idx >= NTOK * DIi) return;
  int n = idx / DIi, c = idx % DIi;
  int bq = n / SEQL, l = n % SEQL;
  float acc = cb[c];
#pragma unroll
  for (int k = 0; k < 4; ++k) {
    int ll = l - 3 + k;
    if (ll >= 0) acc += xz[(size_t)(bq * SEQL + ll) * (2 * DIi) + c] * cw[c * 4 + k];
  }
  float v = siluf_(acc);
  u[idx] = v;
  u_h[idx] = f2h(v);
}

// ---------------------------------------------------------------------------
// Fused selective scan: dt-proj(+softplus) + SSM recurrence + D*u + SiLU(z)
// One thread per (batch, channel); 16-entry state in registers; shared 64-float
// xdb row staged in LDS per timestep. grid = (DI/256, BS)
// ---------------------------------------------------------------------------
__global__ __launch_bounds__(256) void scan_kernel(const float* __restrict__ xz,
                                                   const float* __restrict__ u,
                                                   const float* __restrict__ xdb,
                                                   const float* __restrict__ Wdt,
                                                   const float* __restrict__ bdt,
                                                   const float* __restrict__ Alog,
                                                   const float* __restrict__ Dv,
                                                   unsigned short* __restrict__ y_h) {
  __shared__ float sx[64];
  const int d = blockIdx.x * 256 + threadIdx.x;
  const int bq = blockIdx.y;

  float wdt[DRr];
#pragma unroll
  for (int r = 0; r < DRr; ++r) wdt[r] = Wdt[(size_t)d * DRr + r];
  float Av[DST], hs[DST];
#pragma unroll
  for (int s = 0; s < DST; ++s) {
    Av[s] = -__expf(Alog[(size_t)d * DST + s]);
    hs[s] = 0.0f;
  }
  const float bd = bdt[d];
  const float Dd = Dv[d];

  for (int t = 0; t < SEQL; ++t) {
    const size_t n = (size_t)bq * SEQL + t;
    if (threadIdx.x < 64) sx[threadIdx.x] = xdb[n * 64 + threadIdx.x];
    __syncthreads();
    float dr = bd;
#pragma unroll
    for (int r = 0; r < DRr; ++r) dr += sx[r] * wdt[r];
    const float dt = softplusf_(dr);
    const float ut = u[n * DIi + d];
    float ys = 0.0f;
#pragma unroll
    for (int s = 0; s < DST; ++s) {
      float dA = __expf(dt * Av[s]);
      hs[s] = dA * hs[s] + dt * sx[DRr + s] * ut;
      ys += hs[s] * sx[DRr + DST + s];
    }
    const float z = xz[n * (2 * DIi) + DIi + d];
    const float y = (ys + Dd * ut) * siluf_(z);
    y_h[n * DIi + d] = f2h(y);
    __syncthreads();
  }
}

// ---------------------------------------------------------------------------
extern "C" void kernel_launch(void* const* d_in, const int* in_sizes, int n_in,
                              void* d_out, int out_size, void* d_ws, size_t ws_size,
                              hipStream_t stream) {
  (void)in_sizes; (void)n_in; (void)out_size; (void)ws_size;
  const float* x      = (const float*)d_in[0];
  const float* W_emb  = (const float*)d_in[1];
  const float* b_emb  = (const float*)d_in[2];
  const float* ln_w   = (const float*)d_in[3];
  const float* ln_b   = (const float*)d_in[4];
  const float* W_in   = (const float*)d_in[5];
  const float* conv_w = (const float*)d_in[6];
  const float* conv_b = (const float*)d_in[7];
  const float* W_xp   = (const float*)d_in[8];
  const float* W_dt   = (const float*)d_in[9];
  const float* b_dt   = (const float*)d_in[10];
  const float* A_log  = (const float*)d_in[11];
  const float* Dv     = (const float*)d_in[12];
  const float* W_out  = (const float*)d_in[13];
  const float* fn_w   = (const float*)d_in[14];
  const float* fn_b   = (const float*)d_in[15];
  const float* W_head = (const float*)d_in[16];

  char* p = (char*)d_ws;
  auto alloc = [&](size_t bytes) -> char* {
    char* r = p;
    p += (bytes + 255) & ~(size_t)255;
    return r;
  };
  float* res           = (float*)alloc((size_t)NTOK * DMm * 4);
  float* h             = (float*)alloc((size_t)NTOK * DMm * 4);
  unsigned short* hn_h = (unsigned short*)alloc((size_t)NTOK * DMm * 2);
  float* xz            = (float*)alloc((size_t)NTOK * 2 * DIi * 4);
  float* u             = (float*)alloc((size_t)NTOK * DIi * 4);
  unsigned short* u_h  = (unsigned short*)alloc((size_t)NTOK * DIi * 2);
  float* xdb           = (float*)alloc((size_t)NTOK * 64 * 4);
  unsigned short* y_h  = (unsigned short*)alloc((size_t)NTOK * DIi * 2);
  unsigned short* Win_h  = (unsigned short*)alloc((size_t)NLl * 2 * DIi * DMm * 2);
  unsigned short* Wxp_h  = (unsigned short*)alloc((size_t)NLl * 64 * DIi * 2);
  unsigned short* Wout_h = (unsigned short*)alloc((size_t)NLl * DMm * DIi * 2);
  unsigned short* Whd_h  = (unsigned short*)alloc((size_t)CCc * DMm * 2);

  const int nWin = NLl * 2 * DIi * DMm;
  const int nWxp = NLl * 64 * DIi;
  const int nWout = NLl * DMm * DIi;
  const int nWhd = CCc * DMm;
  cvt_f32_f16<<<(nWin + 255) / 256, 256, 0, stream>>>(W_in, Win_h, nWin);
  cvt_f32_f16<<<(nWxp + 255) / 256, 256, 0, stream>>>(W_xp, Wxp_h, nWxp);
  cvt_f32_f16<<<(nWout + 255) / 256, 256, 0, stream>>>(W_out, Wout_h, nWout);
  cvt_f32_f16<<<(nWhd + 255) / 256, 256, 0, stream>>>(W_head, Whd_h, nWhd);

  embed_kernel<<<(NTOK * DMm + 255) / 256, 256, 0, stream>>>(x, W_emb, b_emb, h);

  for (int i = 0; i < NLl; ++i) {
    ln_kernel<<<NTOK, 256, 0, stream>>>(res, h, (i > 0) ? 1 : 0, ln_w + i * DMm,
                                        ln_b + i * DMm, hn_h);
    gemm_f16_wmma<<<dim3(2 * DIi / 64, NTOK / 64), 256, 0, stream>>>(
        (const unsigned int*)hn_h, (const unsigned int*)(Win_h + (size_t)i * 2 * DIi * DMm),
        xz, NTOK, DMm, 2 * DIi);
    conv_silu_kernel<<<(NTOK * DIi + 255) / 256, 256, 0, stream>>>(
        xz, conv_w + (size_t)i * DIi * 4, conv_b + (size_t)i * DIi, u, u_h);
    gemm_f16_wmma<<<dim3(1, NTOK / 64), 256, 0, stream>>>(
        (const unsigned int*)u_h, (const unsigned int*)(Wxp_h + (size_t)i * 64 * DIi),
        xdb, NTOK, DIi, 64);
    scan_kernel<<<dim3(DIi / 256, BSZ), 256, 0, stream>>>(
        xz, u, xdb, W_dt + (size_t)i * DIi * DRr, b_dt + (size_t)i * DIi,
        A_log + (size_t)i * DIi * DST, Dv + (size_t)i * DIi, y_h);
    gemm_f16_wmma<<<dim3(DMm / 64, NTOK / 64), 256, 0, stream>>>(
        (const unsigned int*)y_h, (const unsigned int*)(Wout_h + (size_t)i * DMm * DIi),
        h, NTOK, DIi, DMm);
  }

  rms_final_kernel<<<NTOK, 256, 0, stream>>>(res, h, fn_w, fn_b, hn_h);
  gemm_f16_wmma<<<dim3((CCc + 63) / 64, NTOK / 64), 256, 0, stream>>>(
      (const unsigned int*)hn_h, (const unsigned int*)Whd_h, (float*)d_out, NTOK, DMm, CCc);
}
